// ImageLinearAttention_27633819582622
// MI455X (gfx1250) — matrix operand order repaired and checked by reading, not compiled, not run on hardware
//
#include <hip/hip_runtime.h>
#include <hip/hip_bf16.h>

// ImageLinearAttention for MI455X (gfx1250, wave32, WMMA).
// Pipeline: fp32->bf16 weights | fused QKV proj (WMMA) | k-softmax(n) |
// q-softmax(d) | ctx = k_sm @ v^T (WMMA, K=4096) | out = ctx^T @ q_sm (WMMA)
// | final proj Wo @ out (WMMA, fp32 epilogue).
// Workspace requirement: ~259 MiB (bf16 intermediates; mostly L2-resident).
// Round-2 change: loop nests restructured so no WMMA fragment is loop-invariant
// across an outer register-resident loop -> no scratch spills in hot loops.

typedef __bf16 bf16;
typedef __attribute__((ext_vector_type(16))) __bf16 v16bf;
typedef __attribute__((ext_vector_type(8)))  float  v8f;

#define DEV __device__ __forceinline__

// ---- problem constants ----
constexpr int BATCH = 16;
constexpr int CHAN  = 256;
constexpr int HEADS = 8;
constexpr int KD    = 64;    // KEY_DIM == VALUE_DIM
constexpr int QKV   = 512;   // HEADS*KD
constexpr int NPOS  = 4096;  // 64*64
constexpr float QK_SCALE = 0.35355339059327373f; // 64^-0.25

// ---- workspace layout (bytes) ----
constexpr size_t WQOFF = 0;                          // Wq bf16 512x256
constexpr size_t WKOFF = WQOFF + (size_t)QKV*CHAN*2;
constexpr size_t WVOFF = WKOFF + (size_t)QKV*CHAN*2;
constexpr size_t WOOFF = WVOFF + (size_t)QKV*CHAN*2; // Wo bf16 256x512
constexpr size_t QOFF  = 1ull << 20;
constexpr size_t QSZ   = (size_t)BATCH*QKV*NPOS*2;   // 64 MiB each
constexpr size_t KOFF  = QOFF + QSZ;
constexpr size_t VOFF  = KOFF + QSZ;
constexpr size_t CTOFF = VOFF + QSZ;                 // ctx^T bf16 [b][h][e][d]
constexpr size_t CTSZ  = (size_t)BATCH*HEADS*KD*KD*2;
constexpr size_t AOOFF = CTOFF + CTSZ;               // attn out bf16 [b][512][n]

// ---------------- WMMA fragment helpers (bf16 16x16x32) ----------------
union FragU { v16bf v; uint4 q[2]; };

// A fragment: 16x32 (MxK), source row-major with K contiguous, row stride ldk.
DEV v16bf load_frag_a(const bf16* base, int ldk, int lane) {
  const int row  = lane & 15;
  const int koff = (lane >> 4) * 8;
  const bf16* p = base + row * ldk + koff;
  FragU f;
  f.q[0] = *(const uint4*)(p);        // K = koff .. koff+7
  f.q[1] = *(const uint4*)(p + 16);   // K = koff+16 .. koff+23
  return f.v;
}

// B fragment: 32x16 (KxN), source is B^T row-major (N rows, K contiguous).
DEV v16bf load_frag_b(const bf16* base, int ldk, int lane) {
  const int col = lane & 15;
  const int kb  = (lane >> 4) * 8;
  const bf16* p = base + col * ldk + kb;
  FragU f;
  f.q[0] = *(const uint4*)(p);
  f.q[1] = *(const uint4*)(p + 16);
  return f.v;
}

DEV v8f wmma_bf16(v16bf a, v16bf b, v8f c) {
  return __builtin_amdgcn_wmma_f32_16x16x32_bf16(
      /*neg_a=*/false, a, /*neg_b=*/false, b,
      /*c_mod=*/(short)0, c, /*reuse_a=*/false, /*reuse_b=*/false);
}

// ---------------- K0: weight fp32 -> bf16 ----------------
__global__ void k_cvt_weights(const float* __restrict__ Wq,
                              const float* __restrict__ Wk,
                              const float* __restrict__ Wv,
                              const float* __restrict__ Wo,
                              bf16* __restrict__ dst) {
  const int i = blockIdx.x * blockDim.x + threadIdx.x;  // 0..524287
  const int M = QKV * CHAN;                             // 131072
  float v;
  if      (i <     M) v = Wq[i];
  else if (i < 2 * M) v = Wk[i -     M];
  else if (i < 3 * M) v = Wv[i - 2 * M];
  else                v = Wo[i - 3 * M];
  dst[i] = (bf16)v;
}

// ---------------- K1: fused QKV projection ----------------
// out[pid][b][m][n] = (sum_c W[m][c]*x[b][c][n] + bias[m]) * scale(pid)
// grid: (ntile=64, proj*mchunk=12, batch=16), block 256 (8 waves).
// Each wave owns ONE 16-row A strip x 4 n-tiles: only loop is kb, nothing
// fragment-shaped is loop-invariant -> no spills.
__global__ void k_qkv_proj(const float* __restrict__ x,
                           const bf16*  __restrict__ wall,
                           const float* __restrict__ bq,
                           const float* __restrict__ bk,
                           const float* __restrict__ bv,
                           bf16* __restrict__ qkvdst) {
  __shared__ bf16 xt[64][264];   // x^T tile [n][c], padded: 528B row, 16B-aligned
  const int tid = threadIdx.x;
  const int nt  = blockIdx.x;
  const int pid = blockIdx.y >> 2;      // 0..2 (q,k,v)
  const int mc  = blockIdx.y & 3;       // 0..3 (m-chunk of 128 rows)
  const int b   = blockIdx.z;
  const int n0  = nt * 64;

  for (int idx = tid; idx < 64 * CHAN; idx += 256) {
    const int n = idx & 63, c = idx >> 6;
    xt[n][c] = (bf16)x[((size_t)b * CHAN + c) * NPOS + n0 + n];
  }
  __syncthreads();

  const bf16*  W    = wall + (size_t)pid * QKV * CHAN;
  const float* bias = (pid == 0) ? bq : (pid == 1) ? bk : bv;
  const float  scale = (pid < 2) ? QK_SCALE : 1.0f;
  bf16* dst = qkvdst + (size_t)pid * BATCH * QKV * NPOS;

  const int lane = tid & 31, wave = tid >> 5;
  const int mt = mc * 8 + wave;                         // 0..31 (512 rows)
  const bf16* Arow = W + (size_t)(mt * 16) * CHAN;

  const v8f zero = {};
  v8f acc[4];
  acc[0] = zero; acc[1] = zero; acc[2] = zero; acc[3] = zero;

  #pragma unroll 1
  for (int kb = 0; kb < CHAN / 32; ++kb) {
    v16bf a = load_frag_a(Arow + kb * 32, CHAN, lane);
    #pragma unroll
    for (int tn = 0; tn < 4; ++tn)
      acc[tn] = wmma_bf16(a, load_frag_b(&xt[tn * 16][kb * 32], 264, lane), acc[tn]);
  }

  const int r0 = (lane >> 4) * 8, col = lane & 15;
  for (int tn = 0; tn < 4; ++tn) {
    #pragma unroll
    for (int r = 0; r < 8; ++r) {
      const int m = mt * 16 + r0 + r;
      const float v = (acc[tn][r] + bias[m]) * scale;
      dst[((size_t)b * QKV + m) * NPOS + n0 + tn * 16 + col] = (bf16)v;
    }
  }
}

// ---------------- K2: k softmax over positions (row of 4096) ----------------
__global__ void k_softmax_k(bf16* __restrict__ kd) {
  __shared__ float red[256];
  const int tid = threadIdx.x;
  bf16* p = kd + (size_t)blockIdx.x * NPOS;
  float v[16];
  float m = -3.0e38f;
  #pragma unroll
  for (int j = 0; j < 16; ++j) { v[j] = (float)p[tid + j * 256]; m = fmaxf(m, v[j]); }
  red[tid] = m; __syncthreads();
  for (int s = 128; s > 0; s >>= 1) {
    if (tid < s) red[tid] = fmaxf(red[tid], red[tid + s]);
    __syncthreads();
  }
  m = red[0]; __syncthreads();
  float sum = 0.f;
  #pragma unroll
  for (int j = 0; j < 16; ++j) { v[j] = __expf(v[j] - m); sum += v[j]; }
  red[tid] = sum; __syncthreads();
  for (int s = 128; s > 0; s >>= 1) {
    if (tid < s) red[tid] += red[tid + s];
    __syncthreads();
  }
  const float inv = 1.0f / red[0];
  #pragma unroll
  for (int j = 0; j < 16; ++j) p[tid + j * 256] = (bf16)(v[j] * inv);
}

// ---------------- K3: q softmax over key-dim (64, stride NPOS) ----------------
__global__ void k_softmax_q(bf16* __restrict__ q) {
  const int n = blockIdx.x * 256 + threadIdx.x;
  const int h = blockIdx.y, b = blockIdx.z;
  bf16* base = q + ((size_t)b * QKV + h * KD) * NPOS + n;
  float v[64];
  float m = -3.0e38f;
  #pragma unroll
  for (int d = 0; d < KD; ++d) { v[d] = (float)base[(size_t)d * NPOS]; m = fmaxf(m, v[d]); }
  float sum = 0.f;
  #pragma unroll
  for (int d = 0; d < KD; ++d) { v[d] = __expf(v[d] - m); sum += v[d]; }
  const float inv = 1.0f / sum;
  #pragma unroll
  for (int d = 0; d < KD; ++d) base[(size_t)d * NPOS] = (bf16)(v[d] * inv);
}

// ---------------- K4: ctx[d][e] = sum_n k_sm[d][n]*v[e][n]; store ctx^T ----
// grid (h=8, b=16), block 256 (8 waves); per wave one 16-row strip, 2 col tiles
__global__ void k_context(const bf16* __restrict__ ks,
                          const bf16* __restrict__ vv,
                          bf16* __restrict__ ctxT) {
  const int h = blockIdx.x, b = blockIdx.y;
  const int tid = threadIdx.x, lane = tid & 31, wave = tid >> 5;
  const int mt  = wave >> 1;            // 0..3  (d tile)
  const int nt0 = (wave & 1) * 2;       // 0 or 2 (e tiles)
  const bf16* ka  = ks + ((size_t)b * QKV + h * KD + mt  * 16) * NPOS;
  const bf16* vb0 = vv + ((size_t)b * QKV + h * KD + nt0 * 16) * NPOS;
  const bf16* vb1 = vb0 + (size_t)16 * NPOS;
  v8f acc0 = {}, acc1 = {};
  #pragma unroll 2
  for (int kb = 0; kb < NPOS / 32; ++kb) {
    v16bf a  = load_frag_a(ka  + kb * 32, NPOS, lane);
    v16bf b0 = load_frag_b(vb0 + kb * 32, NPOS, lane);
    v16bf b1 = load_frag_b(vb1 + kb * 32, NPOS, lane);
    acc0 = wmma_bf16(a, b0, acc0);
    acc1 = wmma_bf16(a, b1, acc1);
    if (kb + 16 < NPOS / 32) {                 // global_prefetch_b8 ahead
      __builtin_prefetch(ka  + (kb + 16) * 32, 0, 0);
      __builtin_prefetch(vb0 + (kb + 16) * 32, 0, 0);
    }
  }
  const int r0 = (lane >> 4) * 8, col = lane & 15;
  bf16* cb = ctxT + (size_t)(b * HEADS + h) * KD * KD;  // [e][d]
  #pragma unroll
  for (int r = 0; r < 8; ++r) {
    const int d = mt * 16 + r0 + r;
    cb[(size_t)(nt0 * 16 + col) * KD + d]       = (bf16)acc0[r];
    cb[(size_t)((nt0 + 1) * 16 + col) * KD + d] = (bf16)acc1[r];
  }
}

// ---------------- K5: out[e][n] = sum_d ctx^T[e][d]*q_sm[d][n] ----------------
// grid (ntile=16 of 256, h=8, b=16), block 256
__global__ void k_apply(const bf16* __restrict__ ctxT,
                        const bf16* __restrict__ q,
                        bf16* __restrict__ ao) {
  __shared__ bf16 qt[256][72];   // q^T tile [n][d], 144B row = 16B-aligned
  const int tid = threadIdx.x, lane = tid & 31, wave = tid >> 5;
  const int n0 = blockIdx.x * 256;
  const int h = blockIdx.y, b = blockIdx.z;
  const bf16* qb = q + ((size_t)b * QKV + h * KD) * NPOS + n0;
  for (int idx = tid; idx < 256 * KD; idx += 256) {
    const int n = idx & 255, d = idx >> 8;
    qt[n][d] = qb[(size_t)d * NPOS + n];
  }
  __syncthreads();
  const bf16* cb = ctxT + (size_t)(b * HEADS + h) * KD * KD;
  const int mt = wave >> 1;                       // e tile 0..3
  v16bf a0 = load_frag_a(cb + (size_t)(mt * 16) * KD,      KD, lane);  // K 0..31
  v16bf a1 = load_frag_a(cb + (size_t)(mt * 16) * KD + 32, KD, lane);  // K 32..63
  bf16* aob = ao + ((size_t)b * QKV + h * KD) * NPOS + n0;
  const int r0 = (lane >> 4) * 8, col = lane & 15;
  #pragma unroll 2
  for (int t = 0; t < 8; ++t) {
    const int ntv = (wave & 1) * 8 + t;           // n tile 0..15
    v8f acc = {};
    acc = wmma_bf16(a0, load_frag_b(&qt[ntv * 16][0],  72, lane), acc);
    acc = wmma_bf16(a1, load_frag_b(&qt[ntv * 16][32], 72, lane), acc);
    #pragma unroll
    for (int r = 0; r < 8; ++r) {
      const int e = mt * 16 + r0 + r;
      aob[(size_t)e * NPOS + ntv * 16 + col] = (bf16)acc[r];
    }
  }
}

// ---------------- K6: final projection Wo[256x512] @ attn + bo -> d_out -----
// grid (ntile=64 of 64, b=16), block 256.
// Nest: kb outer (B frags loaded once per kb), tm inner (A per tm) so the
// B fragments are never invariant across a register-resident loop.
__global__ void k_out_proj(const bf16* __restrict__ ao,
                           const bf16* __restrict__ wo,
                           const float* __restrict__ bo,
                           float* __restrict__ out) {
  __shared__ bf16 at[64][136];   // attn^T K-chunk [n][k], 272B row = 16B-aligned
  const int tid = threadIdx.x, lane = tid & 31, wave = tid >> 5;
  const int n0 = blockIdx.x * 64;
  const int b  = blockIdx.y;
  const v8f zero = {};
  v8f acc[2][4];
  for (int tm = 0; tm < 2; ++tm)
    for (int tn = 0; tn < 4; ++tn) acc[tm][tn] = zero;

  #pragma unroll 1
  for (int kc = 0; kc < 4; ++kc) {               // K chunks of 128
    for (int idx = tid; idx < 64 * 128; idx += 256) {
      const int n = idx & 63, kk = idx >> 6;
      at[n][kk] = ao[((size_t)b * QKV + kc * 128 + kk) * NPOS + n0 + n];
    }
    __syncthreads();
    #pragma unroll 1
    for (int kb = 0; kb < 4; ++kb) {
      v16bf bfr[4];
      #pragma unroll
      for (int tn = 0; tn < 4; ++tn)
        bfr[tn] = load_frag_b(&at[tn * 16][kb * 32], 136, lane);
      #pragma unroll
      for (int tm = 0; tm < 2; ++tm) {
        const int mt = wave * 2 + tm;            // 0..15 (256 rows)
        v16bf a = load_frag_a(wo + (size_t)(mt * 16) * QKV + kc * 128 + kb * 32,
                              QKV, lane);
        #pragma unroll
        for (int tn = 0; tn < 4; ++tn)
          acc[tm][tn] = wmma_bf16(a, bfr[tn], acc[tm][tn]);
      }
    }
    __syncthreads();
  }
  const int r0 = (lane >> 4) * 8, col = lane & 15;
  for (int tm = 0; tm < 2; ++tm) {
    const int mt = wave * 2 + tm;
    for (int tn = 0; tn < 4; ++tn) {
      #pragma unroll
      for (int r = 0; r < 8; ++r) {
        const int m = mt * 16 + r0 + r;
        out[((size_t)b * CHAN + m) * NPOS + n0 + tn * 16 + col] =
            acc[tm][tn][r] + bo[m];
      }
    }
  }
}

// ---------------- launch ----------------
extern "C" void kernel_launch(void* const* d_in, const int* in_sizes, int n_in,
                              void* d_out, int out_size, void* d_ws, size_t ws_size,
                              hipStream_t stream) {
  (void)in_sizes; (void)n_in; (void)out_size; (void)ws_size;
  const float* x  = (const float*)d_in[0];
  const float* Wq = (const float*)d_in[1];
  const float* bq = (const float*)d_in[2];
  const float* Wk = (const float*)d_in[3];
  const float* bk = (const float*)d_in[4];
  const float* Wv = (const float*)d_in[5];
  const float* bv = (const float*)d_in[6];
  const float* Wo = (const float*)d_in[7];
  const float* bo = (const float*)d_in[8];

  char* ws = (char*)d_ws;
  bf16* wall  = (bf16*)(ws + WQOFF);
  bf16* wo_bf = (bf16*)(ws + WOOFF);
  bf16* qkv   = (bf16*)(ws + QOFF);
  bf16* qb    = qkv;
  bf16* kb    = (bf16*)(ws + KOFF);
  bf16* vb    = (bf16*)(ws + VOFF);
  bf16* ctxT  = (bf16*)(ws + CTOFF);
  bf16* ao    = (bf16*)(ws + AOOFF);
  float* out  = (float*)d_out;

  k_cvt_weights<<<2048, 256, 0, stream>>>(Wq, Wk, Wv, Wo, wall);
  k_qkv_proj  <<<dim3(64, 12, BATCH), 256, 0, stream>>>(x, wall, bq, bk, bv, qkv);
  k_softmax_k <<<BATCH * QKV, 256, 0, stream>>>(kb);
  k_softmax_q <<<dim3(16, HEADS, BATCH), 256, 0, stream>>>(qb);
  k_context   <<<dim3(HEADS, BATCH), 256, 0, stream>>>(kb, vb, ctxT);
  k_apply     <<<dim3(16, HEADS, BATCH), 256, 0, stream>>>(ctxT, qb, ao);
  k_out_proj  <<<dim3(64, BATCH), 256, 0, stream>>>(ao, wo_bf, bo, out);
}
